// ModulatedPositionalEmbedder_47090021433734
// MI455X (gfx1250) — matrix operand order repaired and checked by reading, not compiled, not run on hardware
//
#include <hip/hip_runtime.h>
#include <hip/hip_bf16.h>
#include <math.h>

typedef __attribute__((ext_vector_type(16))) _Float16 v16h;
typedef __attribute__((ext_vector_type(8)))  _Float16 v8h;
typedef __attribute__((ext_vector_type(8)))  float    v8f;

#define B_     16
#define N_     4096
#define M_     64
#define L_     16
#define NMOD   512
#define WIDTH_ 256
#define KG     8
#define KI     3

// ---------------------------------------------------------------------------
// Kernel 1: swizzle W0 (128x256) and W1 (256x256) into f16 B-fragment layout.
// Fragment layout [kt][nt][lane][i]: element (lane,i) = W[k][n] with
// k = kt*32 + (i&7) + 8*(lane>=16) + 16*(i>>3), n = nt*16 + (lane&15).   (ISA 7.12.2, B 32x16)
// ---------------------------------------------------------------------------
__global__ void mpe_prep(const float* __restrict__ W0, const float* __restrict__ W1,
                         _Float16* __restrict__ W0h, _Float16* __restrict__ W1h)
{
    int idx = blockIdx.x * blockDim.x + threadIdx.x;
    if (idx < 4 * 16 * 32 * 16) {               // W0: 4 k-tiles of 32
        int i = idx & 15, lane = (idx >> 4) & 31, nt = (idx >> 9) & 15, kt = idx >> 13;
        int k = kt * 32 + (i & 7) + 8 * (lane >> 4) + 16 * (i >> 3);
        int n = nt * 16 + (lane & 15);
        W0h[idx] = (_Float16)W0[k * 256 + n];
    }
    if (idx < 8 * 16 * 32 * 16) {               // W1: 8 k-tiles of 32
        int i = idx & 15, lane = (idx >> 4) & 31, nt = (idx >> 9) & 15, kt = idx >> 13;
        int k = kt * 32 + (i & 7) + 8 * (lane >> 4) + 16 * (i >> 3);
        int n = nt * 16 + (lane & 15);
        W1h[idx] = (_Float16)W1[k * 256 + n];
    }
}

// ---------------------------------------------------------------------------
// Kernel 2: per-batch graph aggregation -> features[b][64][512] (f32, exact).
// ---------------------------------------------------------------------------
__global__ __launch_bounds__(256) void mpe_features(
    const float* __restrict__ pos_z, const float* __restrict__ feat_z,
    const float* __restrict__ Wl, const float* __restrict__ bl,
    const float* __restrict__ Wr, float* __restrict__ features)
{
    __shared__ float fz[M_][L_];
    __shared__ float ag[M_][L_];
    __shared__ float zxs[M_], zys[M_];
    const int b = blockIdx.x;
    const int t = threadIdx.x;

    for (int idx = t; idx < M_ * L_; idx += 256)
        fz[idx >> 4][idx & 15] = feat_z[(size_t)b * M_ * L_ + idx];
    for (int idx = t; idx < M_; idx += 256) {
        zxs[idx] = pos_z[((size_t)b * M_ + idx) * 2 + 0];
        zys[idx] = pos_z[((size_t)b * M_ + idx) * 2 + 1];
    }
    __syncthreads();

    if (t < M_) {
        float px = zxs[t], py = zys[t];
        float bd[KG]; int bi[KG];
#pragma unroll
        for (int k = 0; k < KG; ++k) { bd[k] = 3e38f; bi[k] = 0; }
        for (int j = 0; j < M_; ++j) {          // diag excluded (ref adds 1e10*I)
            if (j == t) continue;
            float dx = px - zxs[j], dy = py - zys[j];
            float d = dx * dx + dy * dy;
            if (d < bd[KG - 1]) {               // stable: ties keep lower index
                int p = KG - 1;
                while (p > 0 && d < bd[p - 1]) { bd[p] = bd[p - 1]; bi[p] = bi[p - 1]; --p; }
                bd[p] = d; bi[p] = j;
            }
        }
#pragma unroll
        for (int l = 0; l < L_; ++l) {
            float s = 0.f;
#pragma unroll
            for (int k = 0; k < KG; ++k) s += fz[bi[k]][l];
            ag[t][l] = s * (1.0f / KG);
        }
    }
    __syncthreads();

    for (int idx = t; idx < M_ * NMOD; idx += 256) {
        int m = idx >> 9, c = idx & 511;
        float acc = bl[c];
#pragma unroll
        for (int l = 0; l < L_; ++l)
            acc = fmaf(ag[m][l], Wl[l * NMOD + c], fmaf(fz[m][l], Wr[l * NMOD + c], acc));
        features[(size_t)b * M_ * NMOD + idx] = acc;
    }
}

// ---------------------------------------------------------------------------
// Kernel 3: swizzle features (64 z-points x 512 channels, per batch) into f16
// B-fragment layout [b][kt<2][nt<32][lane<32][i<16]:
//   element = features[k][c], k = kt*32 + (i&7) + 8*(lane>=16) + 16*(i>>3), c = nt*16+(lane&15)
// ---------------------------------------------------------------------------
__global__ __launch_bounds__(256) void mpe_feat_swizzle(
    const float* __restrict__ features, _Float16* __restrict__ featH)
{
    const int b = blockIdx.x;
    const float* fB = features + (size_t)b * M_ * NMOD;
    _Float16* oB = featH + (size_t)b * (2 * 32 * 32 * 16);
    for (int idx = threadIdx.x; idx < 2 * 32 * 32 * 16; idx += 256) {
        int i = idx & 15, lane = (idx >> 4) & 31, nt = (idx >> 9) & 31, kt = idx >> 14;
        int k = kt * 32 + (i & 7) + 8 * (lane >> 4) + 16 * (i >> 3);
        int c = nt * 16 + (lane & 15);
        oB[idx] = (_Float16)fB[k * NMOD + c];
    }
}

// ---------------------------------------------------------------------------
// Kernel 4: fused kNN-interp (as sparse-A WMMA) + Fourier encoding + WMMA MLP.
// 4 waves / block; wave w owns 16 consecutive g-points (one WMMA M-tile).
// ---------------------------------------------------------------------------
__global__ __launch_bounds__(128) void mpe_main(
    const float* __restrict__ pos_g, const float* __restrict__ pos_z,
    const float* __restrict__ b0, const float* __restrict__ b1,
    const float* __restrict__ W2, const float* __restrict__ b2,
    const _Float16* __restrict__ featH, const _Float16* __restrict__ W0h,
    const _Float16* __restrict__ W1h, float* __restrict__ out)
{
    __shared__ __align__(16) _Float16 h_lds[4][16][WIDTH_];   // 32 KB
    __shared__ float pos_s[4][16][2];
    __shared__ int   knn_i[4][16][KI];
    __shared__ float knn_w[4][16][KI];

    const int wave = threadIdx.x >> 5;
    const int lane = threadIdx.x & 31;
    const int hi   = lane >> 4;
    const int ln   = lane & 15;
    const int b    = blockIdx.x >> 6;                 // 64 blocks per batch
    const int pbase = (blockIdx.x & 63) * 64 + wave * 16;

    // ---- top-3 kNN + normalized inverse-distance weights (one point/lane) ----
    if (lane < 16) {
        int g = pbase + lane;
        float px = pos_g[((size_t)b * N_ + g) * 2 + 0];
        float py = pos_g[((size_t)b * N_ + g) * 2 + 1];
        pos_s[wave][lane][0] = px;
        pos_s[wave][lane][1] = py;
        float d0 = 3e38f, d1 = 3e38f, d2 = 3e38f;
        int   i0 = 0,     i1 = 0,     i2 = 0;
        for (int j = 0; j < M_; ++j) {
            float zx = pos_z[((size_t)b * M_ + j) * 2 + 0];
            float zy = pos_z[((size_t)b * M_ + j) * 2 + 1];
            float dx = px - zx, dy = py - zy;
            float d = dx * dx + dy * dy;
            if (d < d0)      { d2 = d1; i2 = i1; d1 = d0; i1 = i0; d0 = d; i0 = j; }
            else if (d < d1) { d2 = d1; i2 = i1; d1 = d;  i1 = j; }
            else if (d < d2) { d2 = d;  i2 = j; }
        }
        float w0 = 1.f / fmaxf(d0, 1e-16f);
        float w1 = 1.f / fmaxf(d1, 1e-16f);
        float w2 = 1.f / fmaxf(d2, 1e-16f);
        float inv = 1.f / (w0 + w1 + w2);
        knn_i[wave][lane][0] = i0;  knn_w[wave][lane][0] = w0 * inv;
        knn_i[wave][lane][1] = i1;  knn_w[wave][lane][1] = w1 * inv;
        knn_i[wave][lane][2] = i2;  knn_w[wave][lane][2] = w2 * inv;
    }
    __syncthreads();

    // ---- build sparse interpolation matrix Wsp (16x64, 3 nnz/row) directly
    //      in f16 A-fragment layout: M = ln, K = z-point index ----
    v16h wf[2];
    {
        int   j0 = knn_i[wave][ln][0], j1 = knn_i[wave][ln][1], j2 = knn_i[wave][ln][2];
        float w0v = knn_w[wave][ln][0], w1v = knn_w[wave][ln][1], w2v = knn_w[wave][ln][2];
#pragma unroll
        for (int kt = 0; kt < 2; ++kt) {
#pragma unroll
            for (int i = 0; i < 16; ++i) {
                int k = kt * 32 + (i & 7) + 8 * hi + 16 * (i >> 3);
                float w = (k == j0) ? w0v : (k == j1) ? w1v : (k == j2) ? w2v : 0.f;
                wf[kt][i] = (_Float16)w;
            }
        }
    }

    // ---- Fourier features directly in f16 A-fragment layout (16x32 tiles).
    //      k-tiles 0,1 are sin(winded), k-tiles 2,3 are cos of the SAME
    //      arguments at the same element index -> one sincosf per argument. ----
    v16h xf[4];
    {
        float px = pos_s[wave][ln][0], py = pos_s[wave][ln][1];
#pragma unroll
        for (int kt = 0; kt < 2; ++kt) {
#pragma unroll
            for (int i = 0; i < 16; ++i) {
                int kk = kt * 32 + (i & 7) + 8 * hi + 16 * (i >> 3);  // winded idx 0..63
                float band = exp2f((10.0f / 31.0f) * (float)(kk >> 1));
                float wv   = ((kk & 1) ? py : px) * band;
                float sv, cv;
                sincosf(wv, &sv, &cv);
                xf[kt][i]     = (_Float16)sv;   // position[k]      (k = kk)
                xf[kt + 2][i] = (_Float16)cv;   // position[k + 64]
            }
        }
    }

    const v16h* FF  = (const v16h*)featH + (size_t)b * (2 * 32 * 32);
    const v16h* W0f = (const v16h*)W0h;
    const v16h* W1f = (const v16h*)W1h;

    // ---- Layer 0: h = relu(s0 + X @ W0 + b0); s0 = Wsp @ features[:, :256] ----
#pragma unroll 1
    for (int nt = 0; nt < 16; ++nt) {
        int n = nt * 16 + ln;
        float bias = b0[n];
        v8f acc;
#pragma unroll
        for (int r = 0; r < 8; ++r) acc[r] = bias;
#pragma unroll
        for (int kt = 0; kt < 2; ++kt) {
            v16h bf = FF[(kt * 32 + nt) * 32 + lane];
            acc = __builtin_amdgcn_wmma_f32_16x16x32_f16(false, wf[kt], false, bf,
                                                         (short)0, acc, false, false);
        }
#pragma unroll
        for (int kt = 0; kt < 4; ++kt) {
            v16h bf = W0f[(kt * 16 + nt) * 32 + lane];
            acc = __builtin_amdgcn_wmma_f32_16x16x32_f16(false, xf[kt], false, bf,
                                                         (short)0, acc, false, false);
        }
#pragma unroll
        for (int r = 0; r < 8; ++r)
            h_lds[wave][r + 8 * hi][n] = (_Float16)fmaxf(acc[r], 0.f);
    }
    __syncthreads();

    // ---- relayout h through LDS into layer-1 A-fragments (2x b128 DS loads) ----
    v16h af[8];
#pragma unroll
    for (int kt = 0; kt < 8; ++kt) {
        v8h lo = *(const v8h*)&h_lds[wave][ln][kt * 32 + 8 * hi];
        v8h hh = *(const v8h*)&h_lds[wave][ln][kt * 32 + 16 + 8 * hi];
        af[kt] = __builtin_shufflevector(lo, hh, 0, 1, 2, 3, 4, 5, 6, 7,
                                                 8, 9, 10, 11, 12, 13, 14, 15);
    }

    // ---- Layer 1 (+ s1 via Wsp WMMA) + thin layer 2 folded in ----
    float p[8];
#pragma unroll
    for (int r = 0; r < 8; ++r) p[r] = 0.f;
#pragma unroll 1
    for (int nt = 0; nt < 16; ++nt) {
        int n = nt * 16 + ln;
        float bias = b1[n];
        v8f acc;
#pragma unroll
        for (int r = 0; r < 8; ++r) acc[r] = bias;
#pragma unroll
        for (int kt = 0; kt < 2; ++kt) {              // s1: channels 256..511
            v16h bf = FF[(kt * 32 + 16 + nt) * 32 + lane];
            acc = __builtin_amdgcn_wmma_f32_16x16x32_f16(false, wf[kt], false, bf,
                                                         (short)0, acc, false, false);
        }
#pragma unroll
        for (int kt = 0; kt < 8; ++kt) {
            v16h bf = W1f[(kt * 16 + nt) * 32 + lane];
            acc = __builtin_amdgcn_wmma_f32_16x16x32_f16(false, af[kt], false, bf,
                                                         (short)0, acc, false, false);
        }
        float w2v = W2[n];
#pragma unroll
        for (int r = 0; r < 8; ++r)
            p[r] += fmaxf(acc[r], 0.f) * w2v;
    }

    // butterfly reduce across the 16 lanes of each half-wave (n dimension)
#pragma unroll
    for (int s = 1; s < 16; s <<= 1)
#pragma unroll
        for (int r = 0; r < 8; ++r)
            p[r] += __shfl_xor(p[r], s, 32);

    if (ln == 0) {
        float bv = b2[0];
#pragma unroll
        for (int r = 0; r < 8; ++r)
            out[(size_t)b * N_ + pbase + r + 8 * hi] = p[r] + bv;
    }
}

// ---------------------------------------------------------------------------
extern "C" void kernel_launch(void* const* d_in, const int* in_sizes, int n_in,
                              void* d_out, int out_size, void* d_ws, size_t ws_size,
                              hipStream_t stream)
{
    const float* pos_g  = (const float*)d_in[0];
    const float* pos_z  = (const float*)d_in[1];
    const float* feat_z = (const float*)d_in[2];
    const float* Wl     = (const float*)d_in[3];
    const float* bl     = (const float*)d_in[4];
    const float* Wr     = (const float*)d_in[5];
    const float* W0     = (const float*)d_in[6];
    const float* b0     = (const float*)d_in[7];
    const float* W1     = (const float*)d_in[8];
    const float* b1     = (const float*)d_in[9];
    const float* W2     = (const float*)d_in[10];
    const float* b2     = (const float*)d_in[11];
    float* out = (float*)d_out;

    char* ws = (char*)d_ws;
    float*    features = (float*)ws;                                  // 2 MB f32
    _Float16* W0h   = (_Float16*)(ws + (size_t)B_ * M_ * NMOD * 4);   // 64 KB f16
    _Float16* W1h   = W0h + 4 * 16 * 32 * 16;                         // 128 KB f16
    _Float16* featH = W1h + 8 * 16 * 32 * 16;                         // 1 MB f16

    mpe_prep<<<(8 * 16 * 32 * 16 + 255) / 256, 256, 0, stream>>>(W0, W1, W0h, W1h);
    mpe_features<<<B_, 256, 0, stream>>>(pos_z, feat_z, Wl, bl, Wr, features);
    mpe_feat_swizzle<<<B_, 256, 0, stream>>>(features, featH);
    mpe_main<<<(B_ * N_) / 64, 128, 0, stream>>>(pos_g, pos_z, b0, b1, W2, b2,
                                                 featH, W0h, W1h, out);
}
